// GraphAttentionLayer_412316860913
// MI455X (gfx1250) — matrix operand order repaired and checked by reading, not compiled, not run on hardware
//
#include <hip/hip_runtime.h>
#include <cstdint>
#include <cstddef>

// ---------------------------------------------------------------------------
// CDNA5 (gfx1250, wave32) fused GAT layer.  B=8, N=2048, F_in=F_out=256.
// Both GEMMs on V_WMMA_F32_16X16X32_BF16 (bf16 in, f32 accumulate).
// Attention is flash-style with an exact precomputed softmax shift, and the
// packed-h tiles are staged through LDS (async global->LDS when available),
// double buffered, shared by all 16 waves of the workgroup.
// ---------------------------------------------------------------------------

typedef __attribute__((ext_vector_type(16))) __bf16       v16bf;
typedef __attribute__((ext_vector_type(8)))  float        v8f;
typedef __attribute__((ext_vector_type(4)))  float        f4;
typedef __attribute__((ext_vector_type(4)))  unsigned int u4;
typedef __attribute__((ext_vector_type(4)))  int          i4;

// async global->LDS builtin takes (int4 AS1*, int4 AS3*, imm offset, imm cpol)
typedef __attribute__((address_space(1))) i4* gi4p;
typedef __attribute__((address_space(3))) i4* li4p;

#define B_DIM     8
#define N_DIM     2048
#define F_DIM     256
#define NEG_SLOPE 0.01f

#if defined(__has_builtin)
#if __has_builtin(__builtin_amdgcn_global_load_async_to_lds_b128) && \
    __has_builtin(__builtin_amdgcn_s_wait_asynccnt)
#define USE_ASYNC_LDS 1
#endif
#endif
#ifndef USE_ASYNC_LDS
#define USE_ASYNC_LDS 0
#endif

union BfFrag {
    v16bf v;
    u4    q[2];
};

// ---------------------------------------------------------------------------
// Kernel 1: pack W (256x256 f32, [k][f]) into bf16 WMMA B-fragments.
// B-fragment (16-bit, 32x16): lane holds column N = lane&15;
// element e holds K = 16*(lane>>4) + e.   Entry key: (kt, ft, lane).
// ---------------------------------------------------------------------------
__global__ void pack_w_kernel(const float* __restrict__ w,
                              unsigned short* __restrict__ w_pack) {
    const int tid  = blockIdx.x * 256 + threadIdx.x;      // 4096 entries
    const int lane = tid & 31;
    const int ft   = (tid >> 5) & 15;
    const int kt   = tid >> 9;                            // 0..7
    const int li   = lane & 15, hi = lane >> 4;
    BfFrag o;
#pragma unroll
    for (int e = 0; e < 16; ++e) {
        const int k = kt * 32 + hi * 16 + e;
        o.v[e] = (__bf16)w[k * F_DIM + ft * 16 + li];
    }
    u4* dst = (u4*)w_pack;
    dst[tid * 2]     = o.q[0];
    dst[tid * 2 + 1] = o.q[1];
}

// ---------------------------------------------------------------------------
// Kernel 2: pack x (16384x256 f32) into bf16 WMMA A-fragments.
// A-fragment (16-bit, 16x32): lane holds row M = lane&15;
// element e holds K = 16*hi + e.
// ---------------------------------------------------------------------------
__global__ void pack_x_kernel(const float* __restrict__ x,
                              unsigned short* __restrict__ x_pack) {
    const int tid     = blockIdx.x * 256 + threadIdx.x;   // 262144 entries
    const int lane    = tid & 31;
    const int kt      = (tid >> 5) & 7;
    const int rowTile = tid >> 8;                         // 0..1023
    const int li      = lane & 15, hi = lane >> 4;
    const float* xb   = x + ((long)rowTile * 16 + li) * F_DIM + kt * 32 + hi * 16;
    const f4 x0 = *(const f4*)(xb);
    const f4 x1 = *(const f4*)(xb + 4);
    const f4 x2 = *(const f4*)(xb + 8);
    const f4 x3 = *(const f4*)(xb + 12);
    BfFrag o;
#pragma unroll
    for (int c = 0; c < 4; ++c) {
        o.v[c]      = (__bf16)x0[c];
        o.v[4 + c]  = (__bf16)x1[c];
        o.v[8 + c]  = (__bf16)x2[c];
        o.v[12 + c] = (__bf16)x3[c];
    }
    u4* dst = (u4*)x_pack;
    dst[tid * 2]     = o.q[0];
    dst[tid * 2 + 1] = o.q[1];
}

// ---------------------------------------------------------------------------
// Kernel 3: h = x @ W.  One wave per 16x16 output tile, K = 256 (8 WMMAs).
// ---------------------------------------------------------------------------
__global__ void gemm_h_kernel(const unsigned short* __restrict__ x_pack,
                              const unsigned short* __restrict__ w_pack,
                              float* __restrict__ h) {
    const int lane    = threadIdx.x & 31;
    const int waveId  = blockIdx.x * 8 + (threadIdx.x >> 5);   // 16384 tiles
    const int rowTile = waveId >> 4;
    const int ft      = waveId & 15;
    const int li      = lane & 15, hi = lane >> 4;

    v8f acc = {0.f, 0.f, 0.f, 0.f, 0.f, 0.f, 0.f, 0.f};
    const u4* xp = (const u4*)x_pack;
    const u4* wp = (const u4*)w_pack;

#pragma unroll
    for (int kt = 0; kt < 8; ++kt) {
        BfFrag a, b;
        const long aidx = (((long)rowTile * 8 + kt) * 32 + lane) * 2;
        const int  bidx = ((kt * 16 + ft) * 32 + lane) * 2;
        a.q[0] = xp[aidx];
        a.q[1] = xp[aidx + 1];
        b.q[0] = wp[bidx];
        b.q[1] = wp[bidx + 1];
        acc = __builtin_amdgcn_wmma_f32_16x16x32_bf16(
                  false, a.v, false, b.v, (short)0, acc, false, false);
    }
#pragma unroll
    for (int v = 0; v < 8; ++v) {
        const long r = (long)rowTile * 16 + v + 8 * hi;
        h[r * F_DIM + ft * 16 + li] = acc[v];
    }
}

// ---------------------------------------------------------------------------
// Kernel 4: pack h into bf16 WMMA B-fragments keyed (b, jt, ft, lane).
// ---------------------------------------------------------------------------
__global__ void pack_h_kernel(const float* __restrict__ h,
                              unsigned short* __restrict__ h_pack) {
    const int tid  = blockIdx.x * 256 + threadIdx.x;      // 262144 entries
    const int lane = tid & 31;
    const int ft   = (tid >> 5) & 15;
    const int jt   = (tid >> 9) & 63;
    const int b    = tid >> 15;
    const int li   = lane & 15, hi = lane >> 4;
    BfFrag o;
#pragma unroll
    for (int e = 0; e < 16; ++e) {
        const int  j = jt * 32 + hi * 16 + e;
        const long r = (long)b * N_DIM + j;
        o.v[e] = (__bf16)h[r * F_DIM + ft * 16 + li];
    }
    u4* dst = (u4*)h_pack;
    dst[tid * 2]     = o.q[0];
    dst[tid * 2 + 1] = o.q[1];
}

// ---------------------------------------------------------------------------
// Kernel 5: s_src/s_dst row reductions (one wave per row of 256 features).
// ---------------------------------------------------------------------------
__global__ void scores_kernel(const float* __restrict__ h,
                              const float* __restrict__ a,
                              float* __restrict__ s_src,
                              float* __restrict__ s_dst) {
    const int lane = threadIdx.x & 31;
    const int row  = blockIdx.x * 8 + (threadIdx.x >> 5);
    const float* hr = h + (long)row * F_DIM;
    float pa = 0.f, pd = 0.f;
#pragma unroll
    for (int i = 0; i < 8; ++i) {
        const int   f = lane + 32 * i;
        const float v = hr[f];
        pa += v * a[f];
        pd += v * a[F_DIM + f];
    }
#pragma unroll
    for (int m = 16; m >= 1; m >>= 1) {
        pa += __shfl_xor(pa, m, 32);
        pd += __shfl_xor(pd, m, 32);
    }
    if (lane == 0) { s_src[row] = pa; s_dst[row] = pd; }
}

// ---------------------------------------------------------------------------
// Kernel 6: per-batch max of s_dst (8 blocks, one per batch).
// ---------------------------------------------------------------------------
__global__ void dstmax_kernel(const float* __restrict__ s_dst,
                              float* __restrict__ dmax) {
    __shared__ float red[8];
    const int b    = blockIdx.x;
    const int t    = threadIdx.x;
    const int lane = t & 31;
    float m = -__builtin_inff();
#pragma unroll
    for (int i = 0; i < 8; ++i)
        m = fmaxf(m, s_dst[b * N_DIM + t + 256 * i]);
#pragma unroll
    for (int s = 16; s >= 1; s >>= 1)
        m = fmaxf(m, __shfl_xor(m, s, 32));
    if (lane == 0) red[t >> 5] = m;
    __syncthreads();
    if (t == 0) {
        float r = red[0];
#pragma unroll
        for (int i = 1; i < 8; ++i) r = fmaxf(r, red[i]);
        dmax[b] = r;
    }
}

// ---------------------------------------------------------------------------
// Kernel 7: fused flash attention + epilogue.
// 128 blocks x 512 threads (16 waves). Block = 8 query tiles x 2 feature
// halves, one batch. h_pack j-tiles (16 KB) staged into LDS, double
// buffered, via async global->LDS copies; all waves consume via ds_load.
// Fixed softmax shift M_i => no rescale in the hot loop.
// ---------------------------------------------------------------------------
__global__ __launch_bounds__(512)
void gat_attention_kernel(const float* __restrict__ adj,
                          const float* __restrict__ h,
                          const unsigned short* __restrict__ h_pack,
                          const float* __restrict__ s_src,
                          const float* __restrict__ s_dst,
                          const float* __restrict__ dmax,
                          const float* __restrict__ a,
                          const float* __restrict__ bias,
                          const float* __restrict__ beta_p,
                          float* __restrict__ out) {
    __shared__ u4 hbuf[2][1024];                 // 2 x 16 KB j-tiles

    const int tid   = threadIdx.x;
    const int lane  = tid & 31;
    const int wave  = tid >> 5;                  // 0..15
    const int itloc = wave >> 1;                 // query tile within block
    const int fh    = wave & 1;                  // feature half (8 ft each)
    const int b     = blockIdx.x >> 4;
    const int itile = (blockIdx.x & 15) * 8 + itloc;
    const int i0    = itile * 16;
    const int li    = lane & 15, hi = lane >> 4;
    const float a_adj = a[2 * F_DIM];
    const float beta  = beta_p[0];
    const long  rowg  = (long)b * N_DIM + i0 + li;       // A-layout row
    const float ssrc  = s_src[rowg];

    // Exact softmax shift: e_ij <= lrelu(ssrc + max_j sdst + max(a_adj,0)).
    const float bnd = ssrc + dmax[b] + fmaxf(a_adj, 0.f);
    const float Mi  = bnd >= 0.f ? bnd : NEG_SLOPE * bnd;

    const v8f zero = {0.f, 0.f, 0.f, 0.f, 0.f, 0.f, 0.f, 0.f};
    v8f acc[8];
#pragma unroll
    for (int t = 0; t < 8; ++t) acc[t] = zero;
    float l_run = 0.f;

    const u4*  hp  = (const u4*)h_pack;
    const long tb0 = (long)b * 64 * 1024;        // u4 index of this batch's tiles

    // ---- prologue: stage j-tile 0 into hbuf[0] -----------------------------
#if USE_ASYNC_LDS
    __builtin_amdgcn_global_load_async_to_lds_b128(
        (gi4p)(hp + tb0 + tid), (li4p)&hbuf[0][tid], 0, 0);
    __builtin_amdgcn_global_load_async_to_lds_b128(
        (gi4p)(hp + tb0 + tid + 512), (li4p)&hbuf[0][tid + 512], 0, 0);
#else
    hbuf[0][tid]       = hp[tb0 + tid];
    hbuf[0][tid + 512] = hp[tb0 + tid + 512];
#endif

    for (int jt = 0; jt < 64; ++jt) {
        const int cur = jt & 1;
#if USE_ASYNC_LDS
        __builtin_amdgcn_s_wait_asynccnt(0);     // my copies into hbuf[cur] done
#endif
        __syncthreads();                         // tile cur ready; prior reads of
                                                 // hbuf[cur^1] retired
#if USE_ASYNC_LDS
        if (jt < 63) {                           // overlap next copy with math
            const long g = tb0 + (long)(jt + 1) * 1024 + tid;
            __builtin_amdgcn_global_load_async_to_lds_b128(
                (gi4p)(hp + g), (li4p)&hbuf[cur ^ 1][tid], 0, 0);
            __builtin_amdgcn_global_load_async_to_lds_b128(
                (gi4p)(hp + g + 512), (li4p)&hbuf[cur ^ 1][tid + 512], 0, 0);
        }
#else
        u4 nxt0, nxt1;
        if (jt < 63) {
            const long g = tb0 + (long)(jt + 1) * 1024 + tid;
            nxt0 = hp[g];
            nxt1 = hp[g + 512];
        }
#endif
        // ---- scores for this 16x32 tile (A-layout) -------------------------
        const int    jb = jt * 32 + hi * 16;
        const float* ab = adj + rowg * (long)N_DIM + jb;
        const f4 A0 = *(const f4*)(ab);
        const f4 A1 = *(const f4*)(ab + 4);
        const f4 A2 = *(const f4*)(ab + 8);
        const f4 A3 = *(const f4*)(ab + 12);
        const float* db = s_dst + (long)b * N_DIM + jb;
        const f4 D0 = *(const f4*)(db);
        const f4 D1 = *(const f4*)(db + 4);
        const f4 D2 = *(const f4*)(db + 8);
        const f4 D3 = *(const f4*)(db + 12);

        float ev[16];
#pragma unroll
        for (int c = 0; c < 4; ++c) {
            ev[c]      = ssrc + D0[c] + a_adj * A0[c];
            ev[4 + c]  = ssrc + D1[c] + a_adj * A1[c];
            ev[8 + c]  = ssrc + D2[c] + a_adj * A2[c];
            ev[12 + c] = ssrc + D3[c] + a_adj * A3[c];
        }
        BfFrag p;
#pragma unroll
        for (int e = 0; e < 16; ++e) {
            const float z  = ev[e] >= 0.f ? ev[e] : NEG_SLOPE * ev[e];
            const float pe = __expf(z - Mi);
            l_run += pe;
            p.v[e] = (__bf16)pe;
        }

        // ---- P @ h over this wave's 8 feature tiles (LDS-resident) ---------
#pragma unroll
        for (int f8 = 0; f8 < 8; ++f8) {
            const int ftl = fh * 8 + f8;
            BfFrag bfr;
            bfr.q[0] = hbuf[cur][(ftl * 32 + lane) * 2];
            bfr.q[1] = hbuf[cur][(ftl * 32 + lane) * 2 + 1];
            acc[f8] = __builtin_amdgcn_wmma_f32_16x16x32_bf16(
                          false, p.v, false, bfr.v, (short)0, acc[f8],
                          false, false);
        }
#if !USE_ASYNC_LDS
        if (jt < 63) {
            hbuf[cur ^ 1][tid]       = nxt0;
            hbuf[cur ^ 1][tid + 512] = nxt1;
        }
#endif
    }

    // combine the two half-lane partial row sums, broadcast per C-layout row
    l_run += __shfl_xor(l_run, 16, 32);
    float lm[8];
#pragma unroll
    for (int v = 0; v < 8; ++v) lm[v] = __shfl(l_run, v + 8 * hi, 32);

    // epilogue: h' = acc / l ; out = elu(beta*h' + h + bias)
#pragma unroll
    for (int f8 = 0; f8 < 8; ++f8) {
        const int ftl = fh * 8 + f8;
#pragma unroll
        for (int v = 0; v < 8; ++v) {
            const long r   = (long)b * N_DIM + i0 + v + 8 * hi;
            const long idx = r * F_DIM + ftl * 16 + li;
            const float hv = acc[f8][v] / lm[v];
            const float o  = beta * hv + h[idx] + bias[ftl * 16 + li];
            out[idx] = o > 0.f ? o : (__expf(o) - 1.f);
        }
    }
}

// ---------------------------------------------------------------------------
// Launch. Workspace layout (bytes), total ~32.3 MB:
//   0         h       (16 MB f32)
//   16777216  s_src   (64 KB)
//   16842752  s_dst   (64 KB)
//   16908288  dmax    (4 KB pad)
//   16912384  w_pack  (128 KB bf16)
//   17043456  x_pack  (8 MB bf16)
//   25432064  h_pack  (8 MB bf16)
// ---------------------------------------------------------------------------
extern "C" void kernel_launch(void* const* d_in, const int* in_sizes, int n_in,
                              void* d_out, int out_size, void* d_ws, size_t ws_size,
                              hipStream_t stream) {
    (void)in_sizes; (void)n_in; (void)out_size; (void)ws_size;
    const float* adj  = (const float*)d_in[0];
    const float* x    = (const float*)d_in[1];
    const float* w    = (const float*)d_in[2];
    const float* bias = (const float*)d_in[3];
    const float* a    = (const float*)d_in[4];
    const float* beta = (const float*)d_in[5];
    float* out = (float*)d_out;

    char* ws = (char*)d_ws;
    float*          h      = (float*)(ws);
    float*          s_src  = (float*)(ws + 16777216);
    float*          s_dst  = (float*)(ws + 16842752);
    float*          dmax   = (float*)(ws + 16908288);
    unsigned short* w_pack = (unsigned short*)(ws + 16912384);
    unsigned short* x_pack = (unsigned short*)(ws + 17043456);
    unsigned short* h_pack = (unsigned short*)(ws + 25432064);

    pack_w_kernel<<<16, 256, 0, stream>>>(w, w_pack);
    pack_x_kernel<<<1024, 256, 0, stream>>>(x, x_pack);
    gemm_h_kernel<<<2048, 256, 0, stream>>>(x_pack, w_pack, h);
    pack_h_kernel<<<1024, 256, 0, stream>>>(h, h_pack);
    scores_kernel<<<2048, 256, 0, stream>>>(h, a, s_src, s_dst);
    dstmax_kernel<<<8, 256, 0, stream>>>(s_dst, dmax);
    gat_attention_kernel<<<128, 512, 0, stream>>>(adj, h, h_pack, s_src, s_dst,
                                                  dmax, a, bias, beta, out);
}
